// ProcessorBlock_42838003810350
// MI455X (gfx1250) — compile-verified, matrix operand order checked by reading, not run on hardware
//
#include <hip/hip_runtime.h>
#include <hip/hip_bf16.h>

#define H       128
#define NNODE   50000
#define NEDGE   600000
#define EPSLN   1e-5f
#define TILE    64

typedef __attribute__((ext_vector_type(16))) __bf16 bf16x16;
typedef __attribute__((ext_vector_type(8)))  float  f32x8;

#if defined(__AMDGCN__) && \
    __has_builtin(__builtin_amdgcn_global_load_async_to_lds_b128) && \
    __has_builtin(__builtin_amdgcn_s_wait_asynccnt)
#define USE_ASYNC_LDS 1
typedef __attribute__((ext_vector_type(4))) int v4i;
typedef __attribute__((address_space(1))) v4i* g4_ptr;   // global int4*
typedef __attribute__((address_space(3))) v4i* l4_ptr;   // LDS int4*
#else
#define USE_ASYNC_LDS 0
#endif

__device__ __forceinline__ __bf16 bf16_of(float f) { return (__bf16)f; }

// ---------------------------------------------------------------------------
// Pack a (K x 128) fp32 row-major weight matrix into bf16 WMMA B-fragment
// order. Fragment tile (kt,nt): 32 lanes x 16 contiguous elements.
// 16-bit operand layout: lane&15 -> column, lane>>4 selects the K+8 group,
// element e -> K = (e&7) + ((e&8)<<1).
// ---------------------------------------------------------------------------
__global__ void pack_weights(const float* __restrict__ W,
                             unsigned short* __restrict__ out, int K) {
  int t = blockIdx.x * blockDim.x + threadIdx.x;
  if (t >= K * H) return;
  int e    = t & 15;
  int lane = (t >> 4) & 31;
  int tile = t >> 9;          // 512 elems per 32x16 tile
  int nt   = tile & 7;        // 128/16 = 8 column tiles
  int kt   = tile >> 3;
  int g    = lane >> 4;
  int k = kt * 32 + (e & 7) + ((e & 8) << 1) + g * 8;
  int n = nt * 16 + (lane & 15);
  out[t] = __builtin_bit_cast(unsigned short, bf16_of(W[k * H + n]));
}

__global__ void zero_f4(float4* __restrict__ p, long long n4) {
  long long i = blockIdx.x * (long long)blockDim.x + threadIdx.x;
  long long s = gridDim.x * (long long)blockDim.x;
  float4 z = make_float4(0.f, 0.f, 0.f, 0.f);
  for (; i < n4; i += s) p[i] = z;
}

// ---------------------------------------------------------------------------
// Core GEMM: wave computes a 16x128 strip. sA bf16 row-major (lda elems).
// wfrag may point to global (L2-hit) or LDS (async-staged) fragments.
// ---------------------------------------------------------------------------
template <int KT>
__device__ __forceinline__ void wmma_gemm(const __bf16* __restrict__ sA, int lda,
                                          const unsigned short* __restrict__ wfrag,
                                          int lane, int row0, f32x8 acc[8]) {
  const int m  = row0 + (lane & 15);
  const int g8 = (lane >> 4) << 3;
  const bf16x16* wp = reinterpret_cast<const bf16x16*>(wfrag);
  for (int kt = 0; kt < KT; ++kt) {
    bf16x16 a;
#pragma unroll
    for (int e = 0; e < 16; ++e) {
      int k = (kt << 5) + (e & 7) + ((e & 8) << 1) + g8;
      a[e] = sA[m * lda + k];
    }
#pragma unroll
    for (int nt = 0; nt < 8; ++nt) {
      bf16x16 b = wp[(kt * 8 + nt) * 32 + lane];
      acc[nt] = __builtin_amdgcn_wmma_f32_16x16x32_bf16(
          false, a, false, b, (short)0, acc[nt], false, false);
    }
  }
}

__device__ __forceinline__ void zero_acc(f32x8 acc[8]) {
#pragma unroll
  for (int i = 0; i < 8; ++i)
#pragma unroll
    for (int r = 0; r < 8; ++r) acc[i][r] = 0.f;
}

// Row-wise LayerNorm stats fully in registers: per accumulator row r, sum the
// 8 in-lane N-tile values, then butterfly-reduce across the 16 lanes holding
// that row (xor masks 1/2/4/8 stay inside each 16-lane half of the wave).
__device__ __forceinline__ void row_stats(const f32x8 acc[8],
                                          float mu[8], float rs[8]) {
#pragma unroll
  for (int r = 0; r < 8; ++r) {
    float s = 0.f, q = 0.f;
#pragma unroll
    for (int nt = 0; nt < 8; ++nt) { float v = acc[nt][r]; s += v; q += v * v; }
#pragma unroll
    for (int msk = 1; msk < 16; msk <<= 1) {
      s += __shfl_xor(s, msk, 32);
      q += __shfl_xor(q, msk, 32);
    }
    float mm = s * (1.f / H);
    mu[r] = mm;
    rs[r] = rsqrtf(q * (1.f / H) - mm * mm + EPSLN);
  }
}

// LDS layouts (fixed, shared by host launch config and device) -------------
#define LDA_E   386                      // 384 + pad -> conflict-free frags
#define LDA_N   258                      // 256 + pad
#define LDHH    130                      // 128 + pad
#define E_SH_OFF   (TILE * LDA_E * 2)                 // 49408
#define E_W2_OFF   (E_SH_OFF + TILE * LDHH * 2)       // 66048
#define E_DST_OFF  (E_W2_OFF + 32768)                 // 98816
#define E_SMEM     (E_DST_OFF + TILE * 4)             // 99072
#define N_SH_OFF   (TILE * LDA_N * 2)                 // 33024
#define N_W2_OFF   (N_SH_OFF + TILE * LDHH * 2)       // 49664
#define N_SMEM     (N_W2_OFF + 32768)                 // 82432

#if USE_ASYNC_LDS
__device__ __forceinline__ void async_stage_w2(const unsigned short* w2f,
                                               char* smem, int w2off, int tid) {
  const char* g = (const char*)w2f;
  char* l = smem + w2off;
  for (int i = tid * 16; i < 32768; i += 128 * 16) {
    __builtin_amdgcn_global_load_async_to_lds_b128(
        (g4_ptr)(unsigned long long)(const void*)(g + i),
        (l4_ptr)(unsigned int)(unsigned long long)(void*)(l + i), 0, 0);
  }
}
#endif

// ---------------------------------------------------------------------------
// Edge kernel: 64 edges / block, 128 threads (4 waves x 16 rows).
// ---------------------------------------------------------------------------
__global__ __launch_bounds__(128)
void edge_kernel(const float* __restrict__ hn, const float* __restrict__ he,
                 const long long* __restrict__ eidx,
                 const float* __restrict__ b1, const float* __restrict__ g1,
                 const float* __restrict__ be1, const float* __restrict__ b2,
                 const float* __restrict__ gn, const float* __restrict__ bn,
                 const unsigned short* __restrict__ w1f,
                 const unsigned short* __restrict__ w2f,
                 float* __restrict__ out_edge, float* __restrict__ agg) {
  extern __shared__ char smem[];
  __bf16* sA   = (__bf16*)smem;
  __bf16* sH   = (__bf16*)(smem + E_SH_OFF);
  int*    sDst = (int*)(smem + E_DST_OFF);

  const int tid  = threadIdx.x;
  const int lane = tid & 31;
  const int wave = tid >> 5;
  const int row0 = wave << 4;
  const int nloc = lane & 15;
  const int g8   = (lane >> 4) << 3;
  const long long e0 = (long long)blockIdx.x * TILE;

#if USE_ASYNC_LDS
  async_stage_w2(w2f, smem, E_W2_OFF, tid);   // overlaps with gather below
  const unsigned short* w2src = (const unsigned short*)(smem + E_W2_OFF);
#else
  const unsigned short* w2src = w2f;
#endif

  // ---- gather [hn[src] | hn[dst] | he] -> bf16 LDS -------------------------
  {
    const int row  = tid & 63;
    const int part = tid >> 6;
    const long long e   = e0 + row;
    const long long src = eidx[e];
    const long long dst = eidx[(long long)NEDGE + e];
    if (part == 0) sDst[row] = (int)dst;
    const float* pn = hn + (size_t)(part == 0 ? src : dst) * H;
    const int cb = part * H;
#pragma unroll 8
    for (int j4 = 0; j4 < 32; ++j4) {
      float4 v = ((const float4*)pn)[j4];
      __bf16* q = sA + row * LDA_E + cb + j4 * 4;
      q[0] = bf16_of(v.x); q[1] = bf16_of(v.y);
      q[2] = bf16_of(v.z); q[3] = bf16_of(v.w);
    }
    const float* pe = he + (size_t)e * H + part * 64;
#pragma unroll 8
    for (int j4 = 0; j4 < 16; ++j4) {
      float4 v = ((const float4*)pe)[j4];
      __bf16* q = sA + row * LDA_E + 256 + part * 64 + j4 * 4;
      q[0] = bf16_of(v.x); q[1] = bf16_of(v.y);
      q[2] = bf16_of(v.z); q[3] = bf16_of(v.w);
    }
  }
#if USE_ASYNC_LDS
  __builtin_amdgcn_s_wait_asynccnt(0);
#endif
  __syncthreads();   // the only block-wide barrier

  // ---- GEMM1 (64x384)x(384x128), bias, register LN + SiLU -> bf16 H-tile --
  f32x8 acc[8];
  zero_acc(acc);
  wmma_gemm<12>(sA, LDA_E, w1f, lane, row0, acc);

#pragma unroll
  for (int nt = 0; nt < 8; ++nt) {
    const float bv = b1[nt * 16 + nloc];
#pragma unroll
    for (int r = 0; r < 8; ++r) acc[nt][r] += bv;
  }
  float mu[8], rs[8];
  row_stats(acc, mu, rs);
#pragma unroll
  for (int nt = 0; nt < 8; ++nt) {
    const int col = nt * 16 + nloc;
    const float gv = g1[col], bev = be1[col];
#pragma unroll
    for (int r = 0; r < 8; ++r) {
      float y = (acc[nt][r] - mu[r]) * rs[r] * gv + bev;
      float h = y / (1.f + __expf(-y));                 // SiLU
      sH[(row0 + r + g8) * LDHH + col] = bf16_of(h);
    }
  }
  // no barrier: this wave's GEMM2 reads only its own 16 H-rows (DScnt orders)

  // ---- GEMM2 (64x128)x(128x128) + bias + residual -------------------------
  f32x8 acc2[8];
  zero_acc(acc2);
  wmma_gemm<4>(sH, LDHH, w2src, lane, row0, acc2);

#pragma unroll
  for (int nt = 0; nt < 8; ++nt) {
    const int col = nt * 16 + nloc;
    const float bv = b2[col];
#pragma unroll
    for (int r = 0; r < 8; ++r)
      acc2[nt][r] += bv + he[(size_t)(e0 + row0 + r + g8) * H + col];
  }

  // ---- final LN in registers, write h_edge_new + scatter to agg -----------
  float mu2[8], rs2[8];
  row_stats(acc2, mu2, rs2);
  int dstv[8];
#pragma unroll
  for (int r = 0; r < 8; ++r) dstv[r] = sDst[row0 + r + g8];
#pragma unroll
  for (int nt = 0; nt < 8; ++nt) {
    const int col = nt * 16 + nloc;
    const float gv = gn[col], bv = bn[col];
#pragma unroll
    for (int r = 0; r < 8; ++r) {
      float y = (acc2[nt][r] - mu2[r]) * rs2[r] * gv + bv;
      out_edge[(size_t)(e0 + row0 + r + g8) * H + col] = y;
      atomicAdd(&agg[(size_t)dstv[r] * H + col], y);    // segment_sum
    }
  }
}

// ---------------------------------------------------------------------------
// Node kernel: 64 nodes / block, input = [h_node | agg] (K = 256).
// ---------------------------------------------------------------------------
__global__ __launch_bounds__(128)
void node_kernel(const float* __restrict__ hn, const float* __restrict__ agg,
                 const float* __restrict__ b1, const float* __restrict__ g1,
                 const float* __restrict__ be1, const float* __restrict__ b2,
                 const float* __restrict__ gn, const float* __restrict__ bn,
                 const unsigned short* __restrict__ w1f,
                 const unsigned short* __restrict__ w2f,
                 float* __restrict__ out_node) {
  extern __shared__ char smem[];
  __bf16* sA = (__bf16*)smem;
  __bf16* sH = (__bf16*)(smem + N_SH_OFF);

  const int tid  = threadIdx.x;
  const int lane = tid & 31;
  const int wave = tid >> 5;
  const int row0 = wave << 4;
  const int nloc = lane & 15;
  const int g8   = (lane >> 4) << 3;
  const long long n0 = (long long)blockIdx.x * TILE;

#if USE_ASYNC_LDS
  async_stage_w2(w2f, smem, N_W2_OFF, tid);
  const unsigned short* w2src = (const unsigned short*)(smem + N_W2_OFF);
#else
  const unsigned short* w2src = w2f;
#endif

  // ---- gather [hn | agg] -> bf16 LDS --------------------------------------
  {
    const int row  = tid & 63;
    const int part = tid >> 6;
    const long long n = n0 + row;
    const int cb = part * H;
    if (n < NNODE) {
      const float* p = (part == 0) ? (hn + (size_t)n * H) : (agg + (size_t)n * H);
#pragma unroll 8
      for (int j4 = 0; j4 < 32; ++j4) {
        float4 v = ((const float4*)p)[j4];
        __bf16* q = sA + row * LDA_N + cb + j4 * 4;
        q[0] = bf16_of(v.x); q[1] = bf16_of(v.y);
        q[2] = bf16_of(v.z); q[3] = bf16_of(v.w);
      }
    } else {
      for (int j = 0; j < H; ++j) sA[row * LDA_N + cb + j] = bf16_of(0.f);
    }
  }
#if USE_ASYNC_LDS
  __builtin_amdgcn_s_wait_asynccnt(0);
#endif
  __syncthreads();

  f32x8 acc[8];
  zero_acc(acc);
  wmma_gemm<8>(sA, LDA_N, w1f, lane, row0, acc);

#pragma unroll
  for (int nt = 0; nt < 8; ++nt) {
    const float bv = b1[nt * 16 + nloc];
#pragma unroll
    for (int r = 0; r < 8; ++r) acc[nt][r] += bv;
  }
  float mu[8], rs[8];
  row_stats(acc, mu, rs);
#pragma unroll
  for (int nt = 0; nt < 8; ++nt) {
    const int col = nt * 16 + nloc;
    const float gv = g1[col], bev = be1[col];
#pragma unroll
    for (int r = 0; r < 8; ++r) {
      float y = (acc[nt][r] - mu[r]) * rs[r] * gv + bev;
      float h = y / (1.f + __expf(-y));
      sH[(row0 + r + g8) * LDHH + col] = bf16_of(h);
    }
  }

  f32x8 acc2[8];
  zero_acc(acc2);
  wmma_gemm<4>(sH, LDHH, w2src, lane, row0, acc2);

#pragma unroll
  for (int nt = 0; nt < 8; ++nt) {
    const int col = nt * 16 + nloc;
    const float bv = b2[col];
#pragma unroll
    for (int r = 0; r < 8; ++r) {
      const long long n = n0 + row0 + r + g8;
      float res = (n < NNODE) ? hn[(size_t)n * H + col] : 0.f;
      acc2[nt][r] += bv + res;
    }
  }

  float mu2[8], rs2[8];
  row_stats(acc2, mu2, rs2);
#pragma unroll
  for (int nt = 0; nt < 8; ++nt) {
    const int col = nt * 16 + nloc;
    const float gv = gn[col], bv = bn[col];
#pragma unroll
    for (int r = 0; r < 8; ++r) {
      const long long n = n0 + row0 + r + g8;
      if (n < NNODE) {
        float y = (acc2[nt][r] - mu2[r]) * rs2[r] * gv + bv;
        out_node[(size_t)n * H + col] = y;
      }
    }
  }
}

// ---------------------------------------------------------------------------
extern "C" void kernel_launch(void* const* d_in, const int* in_sizes, int n_in,
                              void* d_out, int out_size, void* d_ws, size_t ws_size,
                              hipStream_t stream) {
  const float*     h_node = (const float*)d_in[0];
  const float*     h_edge = (const float*)d_in[1];
  const long long* eidx   = (const long long*)d_in[2];
  const float* eW1  = (const float*)d_in[3];
  const float* eb1  = (const float*)d_in[4];
  const float* eg1  = (const float*)d_in[5];
  const float* ebe1 = (const float*)d_in[6];
  const float* eW2  = (const float*)d_in[7];
  const float* eb2  = (const float*)d_in[8];
  const float* nW1  = (const float*)d_in[9];
  const float* nb1  = (const float*)d_in[10];
  const float* ng1  = (const float*)d_in[11];
  const float* nbe1 = (const float*)d_in[12];
  const float* nW2  = (const float*)d_in[13];
  const float* nb2  = (const float*)d_in[14];
  const float* eng  = (const float*)d_in[15];
  const float* enb  = (const float*)d_in[16];
  const float* nng  = (const float*)d_in[17];
  const float* nnb  = (const float*)d_in[18];

  // workspace: agg (N*H f32) then packed bf16 weight fragments
  char* ws = (char*)d_ws;
  float* agg = (float*)ws;
  size_t off = (size_t)NNODE * H * sizeof(float);
  unsigned short* w1e = (unsigned short*)(ws + off); off += (size_t)384 * H * 2;
  unsigned short* w2e = (unsigned short*)(ws + off); off += (size_t)128 * H * 2;
  unsigned short* w1n = (unsigned short*)(ws + off); off += (size_t)256 * H * 2;
  unsigned short* w2n = (unsigned short*)(ws + off);

  float* out_node = (float*)d_out;                       // (N, H) first
  float* out_edge = out_node + (size_t)NNODE * H;        // then (E2, H)

  pack_weights<<<(384 * H + 255) / 256, 256, 0, stream>>>(eW1, w1e, 384);
  pack_weights<<<(128 * H + 255) / 256, 256, 0, stream>>>(eW2, w2e, 128);
  pack_weights<<<(256 * H + 255) / 256, 256, 0, stream>>>(nW1, w1n, 256);
  pack_weights<<<(128 * H + 255) / 256, 256, 0, stream>>>(nW2, w2n, 128);
  zero_f4<<<2048, 256, 0, stream>>>((float4*)agg, (long long)NNODE * H / 4);

  edge_kernel<<<NEDGE / TILE, 128, E_SMEM, stream>>>(
      h_node, h_edge, eidx, eb1, eg1, ebe1, eb2, eng, enb, w1e, w2e,
      out_edge, agg);
  node_kernel<<<(NNODE + TILE - 1) / TILE, 128, N_SMEM, stream>>>(
      h_node, agg, nb1, ng1, nbe1, nb2, nng, nnb, w1n, w2n, out_node);
}
